// SamVisionEncoder_51342039056570
// MI455X (gfx1250) — matrix operand order repaired and checked by reading, not compiled, not run on hardware
//
#include <hip/hip_runtime.h>
#include <hip/hip_bf16.h>
#include <math.h>

// ---------------------------------------------------------------------------
// SAM vision encoder for gfx1250 (MI455X). All GEMM-shaped work runs through
// v_wmma_f32_16x16x32_bf16 (wave32 WMMA). Attention is flash-style (16q x
// 32k blocks, online softmax) so the 4096x4096 global score matrix is never
// materialized. Weights are converted once per launch to bf16 [N,K] layout so
// B-fragments are single contiguous 32B loads.
// ---------------------------------------------------------------------------

typedef __bf16 bf16;
typedef __attribute__((ext_vector_type(16))) __bf16 v16bf;
typedef __attribute__((ext_vector_type(8)))  __bf16 v8bf;
typedef __attribute__((ext_vector_type(8)))  float  v8f;

#define NTOK 4096
#define DMODEL 768
#define NHEADS 12

__device__ inline v16bf ld_a(const bf16* p) {
  // A-fragment per-lane load in this file's contiguous k order (k = 16h + s): 16 contiguous halves
  // at +0 and +8 halves (caller already applied the lane-half +16 shift).
  v8bf lo = *(const v8bf*)p;
  v8bf hi = *(const v8bf*)(p + 8);
  v16bf r;
#pragma unroll
  for (int i = 0; i < 8; ++i) { r[i] = lo[i]; r[i + 8] = hi[i]; }
  return r;
}

__device__ inline v8f wmma_bf(v16bf a, v16bf b, v8f c) {
  return __builtin_amdgcn_wmma_f32_16x16x32_bf16(false, a, false, b, (short)0, c, false, false);
}

__device__ inline float gelu_exact(float x) {
  return 0.5f * x * (1.0f + erff(x * 0.70710678118654752440f));
}

// ---------------------------------------------------------------------------
// Generic bf16 GEMM: out[M,N] = A[M,K] @ WT[N,K]^T (+bias)(+res)(gelu?)
// Block = 256 threads = 8 waves; each wave -> 64(M) x 16(N) tile.
// ---------------------------------------------------------------------------
__global__ __launch_bounds__(256) void k_gemm(
    const bf16* __restrict__ A, const bf16* __restrict__ WT,
    const float* __restrict__ bias, const float* __restrict__ res,
    float* __restrict__ outF, bf16* __restrict__ outB,
    int M, int N, int K, int do_gelu)
{
  int wave = threadIdx.x >> 5;
  int lane = threadIdx.x & 31;
  int lh = lane >> 4;      // lane half (0/1)
  int lm = lane & 15;
  int n0 = blockIdx.x * 16;
  int m0 = (blockIdx.y * 8 + wave) * 64;
  if (m0 >= M || n0 >= N) return;

  v8f c[4] = {};
  const bf16* wrow = WT + (size_t)(n0 + lm) * K + lh * 16;
  const bf16* arow = A + (size_t)(m0 + lm) * K + lh * 16;
  size_t rstride = (size_t)16 * K;

  for (int k = 0; k < K; k += 32) {
    v16bf bfrag = *(const v16bf*)(wrow + k);
#pragma unroll
    for (int t = 0; t < 4; ++t) {
      v16bf afrag = ld_a(arow + (size_t)t * rstride + k);
      c[t] = wmma_bf(afrag, bfrag, c[t]);
    }
  }

  int col = n0 + lm;
  float bv = bias ? bias[col] : 0.0f;
#pragma unroll
  for (int t = 0; t < 4; ++t) {
#pragma unroll
    for (int r = 0; r < 8; ++r) {
      int row = m0 + t * 16 + r + lh * 8;
      float v = c[t][r] + bv;
      if (res) v += res[(size_t)row * N + col];
      if (do_gelu) v = gelu_exact(v);
      if (outF) outF[(size_t)row * N + col] = v;
      if (outB) outB[(size_t)row * N + col] = (bf16)v;
    }
  }
}

// ---------------------------------------------------------------------------
// Weight convert fp32 -> bf16, optionally transposing [K,M] -> [M,K].
// ---------------------------------------------------------------------------
__global__ void k_convert(bf16* __restrict__ dst, const float* __restrict__ src,
                          int Mo, int K, int transposed)
{
  size_t idx = (size_t)blockIdx.x * blockDim.x + threadIdx.x;
  if (idx >= (size_t)Mo * K) return;
  int k = (int)(idx % K);
  int m = (int)(idx / K);
  float v = transposed ? src[(size_t)k * Mo + m] : src[idx];
  dst[idx] = (bf16)v;
}

// ---------------------------------------------------------------------------
// Patch-embed im2col: pixels [3,1024,1024] -> patches [4096, 768] bf16,
// feature order c*256 + iy*16 + ix (matches patch_w [D,3,16,16] flatten).
// ---------------------------------------------------------------------------
__global__ void k_im2col_patch(const float* __restrict__ pix, bf16* __restrict__ dst)
{
  size_t idx = (size_t)blockIdx.x * blockDim.x + threadIdx.x;
  if (idx >= (size_t)NTOK * DMODEL) return;
  int f = (int)(idx % DMODEL);
  int t = (int)(idx / DMODEL);
  int c = f >> 8, r = f & 255, iy = r >> 4, ix = r & 15;
  int py = t >> 6, px = t & 63;
  int Y = (py << 4) + iy, X = (px << 4) + ix;
  dst[idx] = (bf16)pix[(size_t)c * 1048576 + (size_t)Y * 1024 + X];
}

// Neck conv2 3x3 im2col: t1n[4096,256] bf16 -> a2[4096, 2304], feature order
// ci*9 + ky*3 + kx (matches neck_conv2 [O,I,3,3] flatten). Zero pad=1.
__global__ void k_im2col_neck(const bf16* __restrict__ src, bf16* __restrict__ dst)
{
  size_t idx = (size_t)blockIdx.x * blockDim.x + threadIdx.x;
  if (idx >= (size_t)NTOK * 2304) return;
  int f = (int)(idx % 2304);
  int t = (int)(idx / 2304);
  int ci = f / 9;
  int r = f - ci * 9;
  int ky = r / 3, kx = r - ky * 3;
  int y = (t >> 6) + ky - 1, x = (t & 63) + kx - 1;
  bf16 v = (bf16)0.0f;
  if (y >= 0 && y < 64 && x >= 0 && x < 64)
    v = src[(size_t)(y * 64 + x) * 256 + ci];
  dst[idx] = v;
}

// ---------------------------------------------------------------------------
// LayerNorm over C (eps 1e-6). Optional bf16 out, optional fp32 out; transN>0
// writes fp32 channels-first: outF[c*transN + row].
// ---------------------------------------------------------------------------
__global__ __launch_bounds__(256) void k_ln(
    const float* __restrict__ x, const float* __restrict__ w,
    const float* __restrict__ b, bf16* __restrict__ outB,
    float* __restrict__ outF, int C, int transN)
{
  __shared__ float red[256];
  int row = blockIdx.x;
  const float* xr = x + (size_t)row * C;
  float s = 0.0f;
  for (int i = threadIdx.x; i < C; i += 256) s += xr[i];
  red[threadIdx.x] = s; __syncthreads();
  for (int o = 128; o; o >>= 1) { if (threadIdx.x < o) red[threadIdx.x] += red[threadIdx.x + o]; __syncthreads(); }
  float mean = red[0] / (float)C;
  __syncthreads();
  float s2 = 0.0f;
  for (int i = threadIdx.x; i < C; i += 256) { float d = xr[i] - mean; s2 += d * d; }
  red[threadIdx.x] = s2; __syncthreads();
  for (int o = 128; o; o >>= 1) { if (threadIdx.x < o) red[threadIdx.x] += red[threadIdx.x + o]; __syncthreads(); }
  float inv = rsqrtf(red[0] / (float)C + 1e-6f);
  for (int i = threadIdx.x; i < C; i += 256) {
    float v = (xr[i] - mean) * inv * w[i] + b[i];
    if (outB) outB[(size_t)row * C + i] = (bf16)v;
    if (outF) {
      if (transN) outF[(size_t)i * transN + row] = v;
      else outF[(size_t)row * C + i] = v;
    }
  }
}

__global__ void k_cast(bf16* __restrict__ dst, const float* __restrict__ src, size_t n)
{
  size_t idx = (size_t)blockIdx.x * blockDim.x + threadIdx.x;
  if (idx < n) dst[idx] = (bf16)src[idx];
}

// ---------------------------------------------------------------------------
// Pack qkv[token,2304] bf16 into per-(unit=window*head or head) arrays:
// Qh,Kh [units,Lpad,64] row-major, VT [units,64,Lpad]. Window pad tokens take
// the qkv bias (LN output padded with zeros -> qkv = bias); Lpad tail is 0.
// ---------------------------------------------------------------------------
__global__ void k_pack_qkv(const bf16* __restrict__ qkv, const float* __restrict__ qkvb,
                           bf16* __restrict__ Qh, bf16* __restrict__ Kh,
                           bf16* __restrict__ VT, int windowed, int units, int Lpad)
{
  size_t idx = (size_t)blockIdx.x * blockDim.x + threadIdx.x;
  size_t tot = (size_t)units * Lpad * 64;
  if (idx >= tot) return;
  int d = (int)(idx & 63);
  size_t r2 = idx >> 6;
  int tok = (int)(r2 % Lpad);
  int u = (int)(r2 / Lpad);
  int head, src = -1;
  float q = 0.0f, k = 0.0f, v = 0.0f;
  if (windowed) {
    int win = u / NHEADS; head = u - win * NHEADS;
    if (tok < 196) {
      int ty = tok / 14, tx = tok - ty * 14;
      int wy = win / 5, wx = win - wy * 5;
      int y = wy * 14 + ty, x = wx * 14 + tx;
      if (y < 64 && x < 64) src = y * 64 + x;
      else {
        int o = head * 64 + d;
        q = qkvb[o]; k = qkvb[768 + o]; v = qkvb[1536 + o];
      }
    }
  } else { head = u; src = tok; }
  if (src >= 0) {
    const bf16* p = qkv + (size_t)src * 2304 + head * 64 + d;
    q = (float)p[0]; k = (float)p[768]; v = (float)p[1536];
  }
  size_t o = ((size_t)u * Lpad + tok) * 64 + d;
  Qh[o] = (bf16)q; Kh[o] = (bf16)k;
  VT[((size_t)u * 64 + d) * Lpad + tok] = (bf16)v;
}

// ---------------------------------------------------------------------------
// Decomposed rel-pos bias: relH[u,q,kh] = sum_d q[u,q,d]*tabH[qh-kh+S-1,d],
// relW analogous with qw,kw. q uses *unscaled* Q (matches reference).
// ---------------------------------------------------------------------------
__global__ void k_relpos(const bf16* __restrict__ Qh,
                         const float* __restrict__ tabH, const float* __restrict__ tabW,
                         float* __restrict__ relH, float* __restrict__ relW,
                         int units, int Lpad, int S, int L)
{
  size_t idx = (size_t)blockIdx.x * blockDim.x + threadIdx.x;
  size_t tot = (size_t)units * Lpad * S;
  if (idx >= tot) return;
  int kidx = (int)(idx % S);
  size_t r2 = idx / S;
  int q = (int)(r2 % Lpad);
  int u = (int)(r2 / Lpad);
  size_t o = ((size_t)u * Lpad + q) * S + kidx;
  if (q >= L) { relH[o] = 0.0f; relW[o] = 0.0f; return; }
  int qh = q / S, qw = q - qh * S;
  const bf16* qp = Qh + ((size_t)u * Lpad + q) * 64;
  const float* th = tabH + (size_t)(qh - kidx + S - 1) * 64;
  const float* tw = tabW + (size_t)(qw - kidx + S - 1) * 64;
  float sh = 0.0f, sw = 0.0f;
#pragma unroll 8
  for (int d = 0; d < 64; ++d) {
    float qd = (float)qp[d];
    sh += qd * th[d]; sw += qd * tw[d];
  }
  relH[o] = sh; relW[o] = sw;
}

// ---------------------------------------------------------------------------
// Flash attention. One wave handles (unit u, 16-query tile qt); loops key
// blocks of 32. Scores via 4 WMMA (2 key sub-tiles x 2 d-steps), P@V via 4
// WMMA (P transposed through per-wave LDS tile; same-wave LDS is in-order).
// Online softmax in fp32 in the C-fragment row mapping. Writes un-windowed,
// head-concat bf16 [token,768].
// ---------------------------------------------------------------------------
__global__ __launch_bounds__(128) void k_attn(
    const bf16* __restrict__ Qh, const bf16* __restrict__ Kh,
    const bf16* __restrict__ VT, const float* __restrict__ relH,
    const float* __restrict__ relW, bf16* __restrict__ outPack,
    int units, int Lpad, int L, int S, int windowed)
{
  __shared__ bf16 pbuf[4][16 * 32];
  int wv = threadIdx.x >> 5;
  int lane = threadIdx.x & 31;
  int lh = lane >> 4, lm = lane & 15;
  int qtiles = Lpad >> 4;
  int total = units * qtiles;
  int task = blockIdx.x * 4 + wv;
  if (task >= total) task = total - 1;
  int u = task / qtiles;
  int qt = task - u * qtiles;

  const bf16* qbase = Qh + ((size_t)u * Lpad + (qt << 4) + lm) * 64 + lh * 16;
  v16bf aq0 = ld_a(qbase);
  v16bf aq1 = ld_a(qbase + 32);

  float Mr[8], Ls[8];
  v8f acc[4] = {};
#pragma unroll
  for (int r = 0; r < 8; ++r) { Mr[r] = -1e30f; Ls[r] = 0.0f; }
  const float scale = 0.125f;  // 64^-0.5
  bf16* pw = &pbuf[wv][0];
  int nb = Lpad >> 5;

  for (int kb = 0; kb < nb; ++kb) {
    int k0 = kb << 5;
    int key0 = k0 + lm, key1 = k0 + 16 + lm;
    const bf16* kp0 = Kh + ((size_t)u * Lpad + key0) * 64 + lh * 16;
    const bf16* kp1 = Kh + ((size_t)u * Lpad + key1) * 64 + lh * 16;
    v8f s0 = {}, s1 = {};
    s0 = wmma_bf(aq0, *(const v16bf*)kp0, s0);
    s0 = wmma_bf(aq1, *(const v16bf*)(kp0 + 32), s0);
    s1 = wmma_bf(aq0, *(const v16bf*)kp1, s1);
    s1 = wmma_bf(aq1, *(const v16bf*)(kp1 + 32), s1);

    int kh0 = key0 / S, kw0 = key0 - kh0 * S;
    int kh1 = key1 / S, kw1 = key1 - kh1 * S;
    float bm[8], fr[8], rs[8];
#pragma unroll
    for (int r = 0; r < 8; ++r) {
      int q = (qt << 4) + r + lh * 8;
      const float* rh = relH + ((size_t)u * Lpad + q) * S;
      const float* rw = relW + ((size_t)u * Lpad + q) * S;
      float v0 = (key0 < L) ? (s0[r] * scale + rh[kh0] + rw[kw0]) : -1e30f;
      float v1 = (key1 < L) ? (s1[r] * scale + rh[kh1] + rw[kw1]) : -1e30f;
      s0[r] = v0; s1[r] = v1;
      bm[r] = fmaxf(v0, v1);
    }
#pragma unroll
    for (int m = 1; m <= 8; m <<= 1)
#pragma unroll
      for (int r = 0; r < 8; ++r) bm[r] = fmaxf(bm[r], __shfl_xor(bm[r], m, 32));
#pragma unroll
    for (int r = 0; r < 8; ++r) {
      float nm = fmaxf(Mr[r], bm[r]);
      fr[r] = expf(Mr[r] - nm);
      Mr[r] = nm;
      float p0 = expf(s0[r] - nm), p1 = expf(s1[r] - nm);
      s0[r] = p0; s1[r] = p1;
      rs[r] = p0 + p1;
    }
#pragma unroll
    for (int m = 1; m <= 8; m <<= 1)
#pragma unroll
      for (int r = 0; r < 8; ++r) rs[r] += __shfl_xor(rs[r], m, 32);
#pragma unroll
    for (int r = 0; r < 8; ++r) {
      Ls[r] = Ls[r] * fr[r] + rs[r];
      acc[0][r] *= fr[r]; acc[1][r] *= fr[r];
      acc[2][r] *= fr[r]; acc[3][r] *= fr[r];
      int row = r + lh * 8;
      pw[row * 32 + lm] = (bf16)s0[r];
      pw[row * 32 + 16 + lm] = (bf16)s1[r];
    }
    v16bf ap = ld_a(pw + lm * 32 + lh * 16);
#pragma unroll
    for (int n = 0; n < 4; ++n) {
      const bf16* vp = VT + ((size_t)u * 64 + (n << 4) + lm) * Lpad + k0 + lh * 16;
      acc[n] = wmma_bf(ap, *(const v16bf*)vp, acc[n]);
    }
  }

#pragma unroll
  for (int n = 0; n < 4; ++n) {
#pragma unroll
    for (int r = 0; r < 8; ++r) {
      int q = (qt << 4) + r + lh * 8;
      int d = (n << 4) + lm;
      float val = acc[n][r] / Ls[r];
      if (windowed) {
        if (q < 196) {
          int win = u / NHEADS, head = u - win * NHEADS;
          int ty = q / 14, tx = q - ty * 14;
          int wy = win / 5, wx = win - wy * 5;
          int y = wy * 14 + ty, x = wx * 14 + tx;
          if (y < 64 && x < 64)
            outPack[(size_t)(y * 64 + x) * DMODEL + head * 64 + d] = (bf16)val;
        }
      } else {
        outPack[(size_t)q * DMODEL + u * 64 + d] = (bf16)val;
      }
    }
  }
}

// ---------------------------------------------------------------------------
// Host side
// ---------------------------------------------------------------------------
static inline unsigned cdiv(size_t a, size_t b) { return (unsigned)((a + b - 1) / b); }

struct LayerP {
  const float *ln1w, *ln1b, *qkvw, *qkvb, *projw, *projb, *rph, *rpw,
              *ln2w, *ln2b, *l1w, *l1b, *l2w, *l2b;
};

extern "C" void kernel_launch(void* const* d_in, const int* in_sizes, int n_in,
                              void* d_out, int out_size, void* d_ws, size_t ws_size,
                              hipStream_t stream)
{
  auto in = [&](int i) { return (const float*)d_in[i]; };
  const float *pix, *patch_w, *patch_b, *pos_embed;
  const float *neck1, *nl1w, *nl1b, *neck2, *nl2w, *nl2b;
  LayerP LP[4];

  if (in_sizes[0] == 3 * 1024 * 1024) {
    // dict insertion order: pixel_values, then params in construction order
    pix = in(0); patch_w = in(1); patch_b = in(2); pos_embed = in(3);
    neck1 = in(4); nl1w = in(5); nl1b = in(6);
    neck2 = in(7); nl2w = in(8); nl2b = in(9);
    for (int i = 0; i < 4; ++i) {
      int b = 10 + i * 14;
      LP[i].ln1w = in(b + 0);  LP[i].ln1b = in(b + 1);
      LP[i].qkvw = in(b + 2);  LP[i].qkvb = in(b + 3);
      LP[i].projw = in(b + 4); LP[i].projb = in(b + 5);
      LP[i].rph = in(b + 6);   LP[i].rpw = in(b + 7);
      LP[i].ln2w = in(b + 8);  LP[i].ln2b = in(b + 9);
      LP[i].l1w = in(b + 10);  LP[i].l1b = in(b + 11);
      LP[i].l2w = in(b + 12);  LP[i].l2b = in(b + 13);
    }
  } else {
    // jax tree_flatten alphabetical order: params first (layers, neck_*,
    // patch_*, pos_embed), pixel_values last
    for (int i = 0; i < 4; ++i) {
      int b = i * 14;
      LP[i].l1b = in(b + 0);   LP[i].l1w = in(b + 1);
      LP[i].l2b = in(b + 2);   LP[i].l2w = in(b + 3);
      LP[i].ln1b = in(b + 4);  LP[i].ln1w = in(b + 5);
      LP[i].ln2b = in(b + 6);  LP[i].ln2w = in(b + 7);
      LP[i].projb = in(b + 8); LP[i].projw = in(b + 9);
      LP[i].qkvb = in(b + 10); LP[i].qkvw = in(b + 11);
      LP[i].rph = in(b + 12);  LP[i].rpw = in(b + 13);
    }
    neck1 = in(56); neck2 = in(57);
    nl1b = in(58); nl1w = in(59); nl2b = in(60); nl2w = in(61);
    patch_b = in(62); patch_w = in(63); pos_embed = in(64); pix = in(65);
  }

  // ---- workspace layout ----
  unsigned char* base = (unsigned char*)d_ws;
  size_t cur = 0;
  auto alloc = [&](size_t bytes) {
    void* p = (void*)(base + cur);
    cur += (bytes + 255) & ~(size_t)255;
    return p;
  };
  float* x        = (float*)alloc((size_t)NTOK * DMODEL * 4);
  bf16*  xn       = (bf16*)alloc((size_t)NTOK * DMODEL * 2);    // also patch im2col
  bf16*  qkv      = (bf16*)alloc((size_t)NTOK * 2304 * 2);      // also neck im2col
  bf16*  attnp    = (bf16*)alloc((size_t)NTOK * DMODEL * 2);
  // Qh/Kh/VT region, reused by MLP hidden (25.8MB >= 25.2MB)
  unsigned char* region = (unsigned char*)alloc(3 * (size_t)8601600);
  bf16* Qh = (bf16*)region;
  bf16* Kh = (bf16*)(region + 8601600);
  bf16* VT = (bf16*)(region + 2 * (size_t)8601600);
  bf16* hbuf = (bf16*)region;                                    // [4096,3072] bf16
  float* relH = (float*)alloc((size_t)12 * 4096 * 64 * 4);
  float* relW = (float*)alloc((size_t)12 * 4096 * 64 * 4);
  bf16* wq  = (bf16*)alloc((size_t)2304 * 768 * 2);
  bf16* wp  = (bf16*)alloc((size_t)768 * 768 * 2);
  bf16* w1  = (bf16*)alloc((size_t)3072 * 768 * 2);
  bf16* w2  = (bf16*)alloc((size_t)768 * 3072 * 2);
  bf16* wpa = (bf16*)alloc((size_t)768 * 768 * 2);
  bf16* wn1 = (bf16*)alloc((size_t)256 * 768 * 2);
  bf16* wn2 = (bf16*)alloc((size_t)256 * 2304 * 2);
  float* t1  = (float*)alloc((size_t)NTOK * 256 * 4);
  bf16*  t1n = (bf16*)alloc((size_t)NTOK * 256 * 2);
  float* t2  = (float*)alloc((size_t)NTOK * 256 * 4);
  (void)ws_size; (void)out_size; (void)n_in;

  auto conv = [&](bf16* dst, const float* src, int Mo, int K, int t) {
    k_convert<<<cdiv((size_t)Mo * K, 256), 256, 0, stream>>>(dst, src, Mo, K, t);
  };
  auto gemm = [&](const bf16* A, const bf16* WT_, const float* bias,
                  const float* res, float* oF, bf16* oB,
                  int M, int N, int K, int gelu) {
    dim3 g(N / 16, M / 512);
    k_gemm<<<g, 256, 0, stream>>>(A, WT_, bias, res, oF, oB, M, N, K, gelu);
  };

  // ---- static weights ----
  conv(wpa, patch_w, 768, 768, 0);     // [D,3,16,16] already [out,K]
  conv(wn1, neck1, 256, 768, 0);       // [256,768,1,1]
  conv(wn2, neck2, 256, 2304, 0);      // [256,256,3,3]

  // ---- patch embed + pos ----
  k_im2col_patch<<<cdiv((size_t)NTOK * DMODEL, 256), 256, 0, stream>>>(pix, xn);
  gemm(xn, wpa, patch_b, pos_embed, x, nullptr, NTOK, DMODEL, 768, 0);

  // ---- transformer layers ----
  for (int i = 0; i < 4; ++i) {
    int windowed = (i != 3);
    int units = windowed ? 25 * NHEADS : NHEADS;
    int Lpad  = windowed ? 224 : 4096;
    int L     = windowed ? 196 : 4096;
    int S     = windowed ? 14 : 64;

    conv(wq, LP[i].qkvw, 2304, 768, 1);   // [768,2304] -> [2304,768]
    conv(wp, LP[i].projw, 768, 768, 1);
    conv(w1, LP[i].l1w, 3072, 768, 1);
    conv(w2, LP[i].l2w, 768, 3072, 1);

    k_ln<<<NTOK, 256, 0, stream>>>(x, LP[i].ln1w, LP[i].ln1b, xn, nullptr, DMODEL, 0);
    gemm(xn, wq, LP[i].qkvb, nullptr, nullptr, qkv, NTOK, 2304, 768, 0);

    size_t packN = (size_t)units * Lpad * 64;
    k_pack_qkv<<<cdiv(packN, 256), 256, 0, stream>>>(qkv, LP[i].qkvb, Qh, Kh, VT,
                                                     windowed, units, Lpad);
    size_t relN = (size_t)units * Lpad * S;
    k_relpos<<<cdiv(relN, 256), 256, 0, stream>>>(Qh, LP[i].rph, LP[i].rpw,
                                                  relH, relW, units, Lpad, S, L);
    int tasks = units * (Lpad >> 4);
    k_attn<<<cdiv((size_t)tasks, 4), 128, 0, stream>>>(Qh, Kh, VT, relH, relW, attnp,
                                                       units, Lpad, L, S, windowed);
    gemm(attnp, wp, LP[i].projb, x, x, nullptr, NTOK, DMODEL, 768, 0);

    k_ln<<<NTOK, 256, 0, stream>>>(x, LP[i].ln2w, LP[i].ln2b, xn, nullptr, DMODEL, 0);
    gemm(xn, w1, LP[i].l1b, nullptr, nullptr, hbuf, NTOK, 3072, 768, 1);
    gemm(hbuf, w2, LP[i].l2b, x, x, nullptr, NTOK, DMODEL, 3072, 0);
  }

  // ---- neck ----
  k_cast<<<cdiv((size_t)NTOK * DMODEL, 256), 256, 0, stream>>>(xn, x, (size_t)NTOK * DMODEL);
  gemm(xn, wn1, nullptr, nullptr, t1, nullptr, NTOK, 256, 768, 0);
  k_ln<<<NTOK, 256, 0, stream>>>(t1, nl1w, nl1b, t1n, nullptr, 256, 0);
  k_im2col_neck<<<cdiv((size_t)NTOK * 2304, 256), 256, 0, stream>>>(t1n, qkv);
  gemm(qkv, wn2, nullptr, nullptr, t2, nullptr, NTOK, 256, 2304, 0);
  k_ln<<<NTOK, 256, 0, stream>>>(t2, nl2w, nl2b, nullptr, (float*)d_out, 256, NTOK);
}
